// DeepSeekMLAAttention_59176059404994
// MI455X (gfx1250) — matrix operand order repaired and checked by reading, not compiled, not run on hardware
//
#include <hip/hip_runtime.h>

// ---------------- problem constants ----------------
constexpr int Bsz  = 2;
constexpr int Sl   = 2048;
constexpr int HID  = 2048;
constexpr int INs  = 2 * HID;     // 4096
constexpr int Hh   = 16;
constexpr int NOPE = 128;
constexpr int ROPE = 64;
constexpr int VHD  = 128;
constexpr int QKD  = NOPE + ROPE; // 192
constexpr int QR   = 1536;
constexpr int KVR  = 512;
constexpr int KVCP = 640;         // kvc width padded 576 -> 640 (full GEMM tiles)
constexpr int ROWS = Bsz * Sl;    // 4096

typedef __attribute__((ext_vector_type(16))) __bf16        v16bf;
typedef __attribute__((ext_vector_type(8)))  float         v8f;

__device__ __forceinline__ unsigned short f32_to_bf16(float f) {
    unsigned int u = __float_as_uint(f);
    unsigned int r = u + 0x7FFFu + ((u >> 16) & 1u);   // round-to-nearest-even
    return (unsigned short)(r >> 16);
}

__device__ __forceinline__ v8f wmma_bf16(v16bf a, v16bf b, v8f c) {
    return __builtin_amdgcn_wmma_f32_16x16x32_bf16(false, a, false, b, (short)0, c, false, false);
}

// CDNA5 async copy: one instruction = 32 lanes x 16B global -> LDS, +1 ASYNCcnt per wave.
__device__ __forceinline__ void async_copy_b128(unsigned int lds_off, const void* g) {
    asm volatile("global_load_async_to_lds_b128 %0, %1, off"
                 :: "v"(lds_off), "v"((unsigned long long)g) : "memory");
}
template <int N>
__device__ __forceinline__ void wait_async() {
    asm volatile("s_wait_asynccnt %0" :: "i"(N) : "memory");
}
__device__ __forceinline__ unsigned int lds_addr_of(const void* p) {
    return (unsigned int)(unsigned long long)p;   // low 32 bits of flat addr = LDS offset
}

// A-fragment (16x32 bf16): lane m = lane&15, half = lane>>4.
// VGPR i<4: K = half*8 + 2i ; i>=4: K = 16 + half*8 + 2(i-4)  (two 16B runs per lane).
__device__ __forceinline__ v16bf load_frag_a(const unsigned short* p, int ld, int lane) {
    int m = lane & 15, half = lane >> 4;
    const unsigned short* r = p + (size_t)m * ld;
    union { unsigned int u[8]; v16bf b; } f;
#pragma unroll
    for (int i = 0; i < 8; ++i) {
        int k = (i < 4) ? (half * 8 + i * 2) : (16 + half * 8 + (i - 4) * 2);
        f.u[i] = *(const unsigned int*)(r + k);
    }
    return f.b;
}

// B-fragment (32x16 bf16) from K-transposed tile T[n][k] (stride ld):
// VGPR i<4: K = half*8 + 2i ; i>=4: K = 16 + half*8 + 2(i-4)  (same as A-fragment).
__device__ __forceinline__ v16bf load_frag_b(const unsigned short* p, int ld, int lane) {
    int n = lane & 15, half = lane >> 4;
    const unsigned short* r = p + (size_t)n * ld + half * 8;
    union { unsigned int u[8]; v16bf b; } f;
#pragma unroll
    for (int i = 0; i < 8; ++i) f.u[i] = *(const unsigned int*)(r + ((i < 4) ? 2 * i : 16 + 2 * (i - 4)));
    return f.b;
}

// ---------------- GEMM: C[M,N] = A[M,K] @ Bt[N,K]^T, bf16 in, f32 out ----------------
// Full tiles only: M%128==0, N%128==0, K%64==0 (nt even). Ping/pong async LDS staging.
#define GTM 128
#define GTN 128
#define GTK 32
#define GLDA 56   // LDS row stride (ushorts): 112B, 16B-aligned, conflict-free frag reads

__global__ __launch_bounds__(256) void gemm_bf16_kernel(
    const unsigned short* __restrict__ A,   // [M,K] bf16
    const unsigned short* __restrict__ Bt,  // [N,K] bf16
    float* __restrict__ C, int M, int N, int K)
{
    __shared__ unsigned short As0[GTM * GLDA];
    __shared__ unsigned short Bs0[GTN * GLDA];
    __shared__ unsigned short As1[GTM * GLDA];
    __shared__ unsigned short Bs1[GTN * GLDA];

    int tid = threadIdx.x;
    int lane = tid & 31, wave = tid >> 5;
    int half = lane >> 4, ln = lane & 15;
    int bm = blockIdx.y * GTM, bn = blockIdx.x * GTN;
    int wm = (wave >> 1) * 32;   // 4 row groups of 32
    int wn = (wave & 1) * 64;    // 2 col groups of 64
    unsigned int a0 = lds_addr_of(As0), b0 = lds_addr_of(Bs0);
    unsigned int a1 = lds_addr_of(As1), b1 = lds_addr_of(Bs1);

    // 1024 B128 lane-transfers per tile = 4 async instructions per wave
    auto issue_tile = [&](unsigned int aBase, unsigned int bBase, int k0) {
#pragma unroll
        for (int it = 0; it < 4; ++it) {
            int t = tid + it * 256;
            int r = (t >> 2) & 127, seg = t & 3;
            bool isB = t >= 512;                 // constant per unrolled it
            const unsigned short* g = (isB ? Bt + (size_t)(bn + r) * K
                                           : A  + (size_t)(bm + r) * K) + k0 + seg * 8;
            unsigned int lds = (isB ? bBase : aBase) + (unsigned)(r * GLDA + seg * 8) * 2;
            async_copy_b128(lds, g);
        }
    };

    v8f acc[2][4] = {};
    auto compute = [&](const unsigned short* AsT, const unsigned short* BsT) {
#pragma unroll
        for (int i = 0; i < 2; ++i) {
            v16bf a = load_frag_a(AsT + (wm + 16 * i) * GLDA, GLDA, lane);
#pragma unroll
            for (int j = 0; j < 4; ++j) {
                v16bf b = load_frag_b(BsT + (wn + 16 * j) * GLDA, GLDA, lane);
                acc[i][j] = wmma_bf16(a, b, acc[i][j]);
            }
        }
    };

    int nt = K / GTK;   // even for every call site
    issue_tile(a0, b0, 0);
    for (int kt = 0; kt < nt; kt += 2) {
        // phase 0: compute buf0, prefetch buf1
        issue_tile(a1, b1, (kt + 1) * GTK);      // kt+1 < nt always (nt even)
        wait_async<4>();
        __syncthreads();
        compute(As0, Bs0);
        __syncthreads();
        // phase 1: compute buf1, prefetch buf0
        if (kt + 2 < nt) { issue_tile(a0, b0, (kt + 2) * GTK); wait_async<4>(); }
        else             { wait_async<0>(); }
        __syncthreads();
        compute(As1, Bs1);
        __syncthreads();
    }

    // epilogue: per-row base pointer + immediate offsets
#pragma unroll
    for (int i = 0; i < 2; ++i) {
        float* base = C + (size_t)(bm + wm + 16 * i + half * 8) * N + (bn + wn + ln);
#pragma unroll
        for (int v = 0; v < 8; ++v) {
            float* rp = base + (size_t)v * N;
#pragma unroll
            for (int j = 0; j < 4; ++j) rp[j * 16] = acc[i][j][v];
        }
    }
}

// ---------------- f32 -> bf16 elementwise (strided rows) ----------------
__global__ __launch_bounds__(256) void conv_bf16_kernel(const float* __restrict__ src,
                                                        unsigned short* __restrict__ dst,
                                                        int rows, int cols, int sstride)
{
    size_t total = (size_t)rows * cols;
    for (size_t i = (size_t)blockIdx.x * 256 + threadIdx.x; i < total;
         i += (size_t)gridDim.x * 256) {
        size_t r = i / (size_t)cols, c = i % (size_t)cols;
        dst[i] = f32_to_bf16(src[r * sstride + c]);
    }
}

// ---------------- f32 [R,C] -> bf16 transpose [CP,R] (zero-pad rows C..CP) ----------------
__global__ __launch_bounds__(256) void transpose_bf16_kernel(const float* __restrict__ src,
                                                             unsigned short* __restrict__ dst,
                                                             int R, int C, int CP)
{
    __shared__ float t[32][33];
    int c0 = blockIdx.x * 32, r0 = blockIdx.y * 32;
    for (int i = threadIdx.y; i < 32; i += 8) {
        int r = r0 + i, c = c0 + (int)threadIdx.x;
        t[i][threadIdx.x] = (r < R && c < C) ? src[(size_t)r * C + c] : 0.f;
    }
    __syncthreads();
    for (int i = threadIdx.y; i < 32; i += 8) {
        int n = c0 + i, k = r0 + (int)threadIdx.x;
        if (n < CP && k < R) dst[(size_t)n * R + k] = f32_to_bf16(t[threadIdx.x][i]);
    }
}

// ---------------- bf16 V [B,H,S,VHD] -> Vt [B,H,VHD,S] ----------------
__global__ __launch_bounds__(256) void transpose_v_kernel(const unsigned short* __restrict__ Vb,
                                                          unsigned short* __restrict__ Vt)
{
    __shared__ unsigned short t[32][33];
    int z = blockIdx.z;                               // b*H + h
    const unsigned short* s = Vb + (size_t)z * Sl * VHD;
    unsigned short*       d = Vt + (size_t)z * VHD * Sl;
    int c0 = blockIdx.x * 32, r0 = blockIdx.y * 32;   // c over VHD, r over S
    for (int i = threadIdx.y; i < 32; i += 8)
        t[i][threadIdx.x] = s[(size_t)(r0 + i) * VHD + c0 + threadIdx.x];
    __syncthreads();
    for (int i = threadIdx.y; i < 32; i += 8)
        d[(size_t)(c0 + i) * Sl + r0 + threadIdx.x] = t[threadIdx.x][i];
}

// ---------------- RMS norm (in place), block per row ----------------
__global__ __launch_bounds__(256) void rms_kernel(float* __restrict__ x,
                                                  const float* __restrict__ w,
                                                  int cols, int stride)
{
    int row = blockIdx.x;
    float* xr = x + (size_t)row * stride;
    __shared__ float red[256];
    float s = 0.f;
    for (int c = threadIdx.x; c < cols; c += 256) { float v = xr[c]; s += v * v; }
    red[threadIdx.x] = s;
    __syncthreads();
    for (int o = 128; o > 0; o >>= 1) {
        if (threadIdx.x < (unsigned)o) red[threadIdx.x] += red[threadIdx.x + o];
        __syncthreads();
    }
    float r = rsqrtf(red[0] / (float)cols + 1e-6f);
    for (int c = threadIdx.x; c < cols; c += 256) xr[c] = xr[c] * r * w[c];
}

// ---------------- RoPE + pack Q/K/V to bf16 ----------------
__global__ __launch_bounds__(256) void rope_pack_kernel(
    const float* __restrict__ q,   // [ROWS, H*QKD]
    const float* __restrict__ kv,  // [ROWS, H*(NOPE+VHD)]
    const float* __restrict__ kvc, // [ROWS, KVCP] (k_rope raw at col 512..575)
    const int* __restrict__ pos_ids,
    unsigned short* __restrict__ Qf, // [B,H,S,QKD] bf16 (pre-scaled by 1/sqrt(QKD))
    unsigned short* __restrict__ Kf, // [B,H,S,QKD] bf16
    unsigned short* __restrict__ Vb) // [B,H,S,VHD] bf16
{
    int bs = blockIdx.x;            // b*S + s
    int b = bs / Sl, s = bs % Sl;
    __shared__ float cs[32], sn[32];
    if (threadIdx.x < 32) {
        float p = (float)pos_ids[bs];
        float inv = __expf(-(float)threadIdx.x * (2.0f / (float)ROPE) * 9.210340371976184f);
        float fr = p * inv;
        cs[threadIdx.x] = __cosf(fr);
        sn[threadIdx.x] = __sinf(fr);
    }
    __syncthreads();
    const float scale = rsqrtf((float)QKD);

    for (int i = threadIdx.x; i < Hh * QKD; i += 256) {
        int h = i / QKD, d = i % QKD;
        const float* qrow = q + (size_t)bs * (Hh * QKD) + h * QKD;
        float val;
        if (d < NOPE) val = qrow[d];
        else {
            int ri = d - NOPE, p2 = ri >> 1;
            float x0 = qrow[NOPE + 2 * p2], x1 = qrow[NOPE + 2 * p2 + 1];
            val = (ri & 1) ? (x1 * cs[p2] + x0 * sn[p2]) : (x0 * cs[p2] - x1 * sn[p2]);
        }
        Qf[(((size_t)b * Hh + h) * Sl + s) * QKD + d] = f32_to_bf16(val * scale);
    }
    for (int i = threadIdx.x; i < Hh * QKD; i += 256) {
        int h = i / QKD, d = i % QKD;
        float val;
        if (d < NOPE) val = kv[(size_t)bs * (Hh * (NOPE + VHD)) + h * (NOPE + VHD) + d];
        else {
            const float* kr = kvc + (size_t)bs * KVCP + KVR;
            int ri = d - NOPE, p2 = ri >> 1;
            float x0 = kr[2 * p2], x1 = kr[2 * p2 + 1];
            val = (ri & 1) ? (x1 * cs[p2] + x0 * sn[p2]) : (x0 * cs[p2] - x1 * sn[p2]);
        }
        Kf[(((size_t)b * Hh + h) * Sl + s) * QKD + d] = f32_to_bf16(val);
    }
    for (int i = threadIdx.x; i < Hh * VHD; i += 256) {
        int h = i / VHD, n = i % VHD;
        float val = kv[(size_t)bs * (Hh * (NOPE + VHD)) + h * (NOPE + VHD) + NOPE + n];
        Vb[(((size_t)b * Hh + h) * Sl + s) * VHD + n] = f32_to_bf16(val);
    }
}

// ---------------- causal flash attention, bf16 WMMA + ping/pong async tiles ----------------
#define KSP 200   // K-tile LDS stride (ushorts): 400B, 16B-aligned, conflict-free
#define VTP 72    // V-tile LDS stride: 144B, 16B-aligned, conflict-free
#define PSP 66    // P-patch stride (manual writes)

__global__ __launch_bounds__(128) void mla_attn_kernel(
    const unsigned short* __restrict__ Qf,   // [B,H,S,QKD]
    const unsigned short* __restrict__ Kf,   // [B,H,S,QKD]
    const unsigned short* __restrict__ Vt,   // [B,H,VHD,S]
    float* __restrict__ Oattn)               // [B*S, H*VHD] f32
{
    __shared__ unsigned short Ks0[64 * KSP];
    __shared__ unsigned short Vs0[128 * VTP];
    __shared__ unsigned short Ks1[64 * KSP];
    __shared__ unsigned short Vs1[128 * VTP];
    __shared__ unsigned short Ps[4][16 * PSP];     // per-wave P patch 16x64

    int qt = blockIdx.x, h = blockIdx.y, b = blockIdx.z;
    int tid = threadIdx.x, lane = tid & 31, w = tid >> 5;
    int half = lane >> 4, ln = lane & 15;
    size_t headQ  = (((size_t)b * Hh + h) * Sl) * QKD;
    size_t headVt = (((size_t)b * Hh + h) * VHD) * Sl;
    int q0 = qt * 64;
    unsigned int k0B = lds_addr_of(Ks0), v0B = lds_addr_of(Vs0);
    unsigned int k1B = lds_addr_of(Ks1), v1B = lds_addr_of(Vs1);

    // K tile: 64x384B = 12 insts/wave ; V tile: 128x128B = 8 insts/wave -> 20 ASYNCcnt/tile
    auto issue_kv = [&](unsigned int ksBase, unsigned int vsBase, int kt) {
        int k0 = kt * 64;
#pragma unroll
        for (int it = 0; it < 12; ++it) {
            int t = tid + it * 128;
            int r = t / 24, seg = t % 24;
            async_copy_b128(ksBase + (unsigned)(r * KSP + seg * 8) * 2,
                            Kf + headQ + (size_t)(k0 + r) * QKD + seg * 8);
        }
#pragma unroll
        for (int it = 0; it < 8; ++it) {
            int t = tid + it * 128;
            int n = t >> 3, seg = t & 7;
            async_copy_b128(vsBase + (unsigned)(n * VTP + seg * 8) * 2,
                            Vt + headVt + (size_t)n * Sl + k0 + seg * 8);
        }
    };

    // Q fragments in registers: 16 rows per wave, 6 K-chunks of 32
    v16bf qf[6];
    {
        const unsigned short* qb = Qf + headQ + (size_t)(q0 + w * 16) * QKD;
#pragma unroll
        for (int kc = 0; kc < 6; ++kc) qf[kc] = load_frag_a(qb + kc * 32, QKD, lane);
    }

    v8f o[8] = {};
    float mst[8], lst[8];
#pragma unroll
    for (int v = 0; v < 8; ++v) { mst[v] = -1e30f; lst[v] = 0.f; }

    auto attn_step = [&](const unsigned short* KsT, const unsigned short* VsT, int kt) {
        int k0 = kt * 64;
        // S = Q K^T
        v8f sa[4] = {};
#pragma unroll
        for (int kc = 0; kc < 6; ++kc) {
#pragma unroll
            for (int j = 0; j < 4; ++j) {
                v16bf bf = load_frag_b(KsT + (j * 16) * KSP + kc * 32, KSP, lane);
                sa[j] = wmma_bf16(qf[kc], bf, sa[j]);
            }
        }
        // online softmax
        int qbase = q0 + w * 16 + half * 8;
        float curmax[8];
#pragma unroll
        for (int v = 0; v < 8; ++v) {
            int gq = qbase + v;
            float mx = -1e30f;
#pragma unroll
            for (int j = 0; j < 4; ++j) {
                int gk = k0 + j * 16 + ln;
                float sv = (gk <= gq) ? sa[j][v] : -1e30f;
                sa[j][v] = sv;
                mx = fmaxf(mx, sv);
            }
            mx = fmaxf(mx, __shfl_xor(mx, 1, 32));
            mx = fmaxf(mx, __shfl_xor(mx, 2, 32));
            mx = fmaxf(mx, __shfl_xor(mx, 4, 32));
            mx = fmaxf(mx, __shfl_xor(mx, 8, 32));
            curmax[v] = mx;
        }
#pragma unroll
        for (int v = 0; v < 8; ++v) {
            float mnew = fmaxf(mst[v], curmax[v]);
            float rs = 0.f;
#pragma unroll
            for (int j = 0; j < 4; ++j) {
                float p = __expf(sa[j][v] - mnew);
                rs += p;
                Ps[w][(half * 8 + v) * PSP + j * 16 + ln] = f32_to_bf16(p);
            }
            rs += __shfl_xor(rs, 1, 32);
            rs += __shfl_xor(rs, 2, 32);
            rs += __shfl_xor(rs, 4, 32);
            rs += __shfl_xor(rs, 8, 32);
            float alpha = __expf(mst[v] - mnew);
            lst[v] = lst[v] * alpha + rs;
            mst[v] = mnew;
#pragma unroll
            for (int j = 0; j < 8; ++j) o[j][v] *= alpha;
        }
        // O += P @ V
#pragma unroll
        for (int kc = 0; kc < 2; ++kc) {
            v16bf a = load_frag_a(&Ps[w][kc * 32], PSP, lane);
#pragma unroll
            for (int j = 0; j < 8; ++j) {
                v16bf bf = load_frag_b(VsT + (j * 16) * VTP + kc * 32, VTP, lane);
                o[j] = wmma_bf16(a, bf, o[j]);
            }
        }
    };

    // ping/pong pipeline over kt = 0..qt (compile-time constant buffer selection)
    issue_kv(k0B, v0B, 0);
    int kt = 0;
    for (;;) {
        // phase 0: compute buf0, prefetch buf1
        if (kt < qt) { issue_kv(k1B, v1B, kt + 1); wait_async<20>(); }
        else         { wait_async<0>(); }
        __syncthreads();
        attn_step(Ks0, Vs0, kt);
        __syncthreads();
        if (kt == qt) break;
        ++kt;
        // phase 1: compute buf1, prefetch buf0
        if (kt < qt) { issue_kv(k0B, v0B, kt + 1); wait_async<20>(); }
        else         { wait_async<0>(); }
        __syncthreads();
        attn_step(Ks1, Vs1, kt);
        __syncthreads();
        if (kt == qt) break;
        ++kt;
    }

    // epilogue: per-row base pointer + immediate offsets
#pragma unroll
    for (int v = 0; v < 8; ++v) {
        float inv = 1.0f / lst[v];
        int gq = q0 + w * 16 + half * 8 + v;
        float* rp = Oattn + ((size_t)b * Sl + gq) * (size_t)(Hh * VHD) + h * VHD + ln;
#pragma unroll
        for (int j = 0; j < 8; ++j) rp[j * 16] = o[j][v] * inv;
    }
}

// ---------------- host side ----------------
extern "C" void kernel_launch(void* const* d_in, const int* in_sizes, int n_in,
                              void* d_out, int out_size, void* d_ws, size_t ws_size,
                              hipStream_t stream) {
    (void)in_sizes; (void)n_in; (void)out_size; (void)ws_size;

    const float* hidden = (const float*)d_in[0];
    const float* Wq_a   = (const float*)d_in[1];
    const float* q_ln   = (const float*)d_in[2];
    const float* Wq_b   = (const float*)d_in[3];
    const float* Wkv_a  = (const float*)d_in[4];
    const float* kv_ln  = (const float*)d_in[5];
    const float* Wkv_b  = (const float*)d_in[6];
    const float* Wo     = (const float*)d_in[7];
    const int*   pos    = (const int*)d_in[8];

    char* ws = (char*)d_ws;
    size_t off = 0;
    auto alloc = [&](size_t bytes) { void* p = ws + off; off += (bytes + 255) & ~(size_t)255; return p; };

    unsigned short* hid_b = (unsigned short*)alloc((size_t)ROWS * INs * 2);
    unsigned short* WqaT  = (unsigned short*)alloc((size_t)QR * INs * 2);
    unsigned short* WqbT  = (unsigned short*)alloc((size_t)Hh * QKD * QR * 2);
    unsigned short* WkvaT = (unsigned short*)alloc((size_t)KVCP * INs * 2);
    unsigned short* WkvbT = (unsigned short*)alloc((size_t)Hh * (NOPE + VHD) * KVR * 2);
    unsigned short* WoT   = (unsigned short*)alloc((size_t)HID * Hh * VHD * 2);
    float* qa_f  = (float*)alloc((size_t)ROWS * QR * 4);
    unsigned short* qa_b = (unsigned short*)alloc((size_t)ROWS * QR * 2);
    float* q_f   = (float*)alloc((size_t)ROWS * Hh * QKD * 4);
    float* kvc_f = (float*)alloc((size_t)ROWS * KVCP * 4);
    unsigned short* kvn_b = (unsigned short*)alloc((size_t)ROWS * KVR * 2);
    float* kv_f  = (float*)alloc((size_t)ROWS * Hh * (NOPE + VHD) * 4);
    unsigned short* Qf  = (unsigned short*)alloc((size_t)Bsz * Hh * Sl * QKD * 2);
    unsigned short* Kf  = (unsigned short*)alloc((size_t)Bsz * Hh * Sl * QKD * 2);
    unsigned short* Vb  = (unsigned short*)alloc((size_t)Bsz * Hh * Sl * VHD * 2);
    unsigned short* Vtg = (unsigned short*)alloc((size_t)Bsz * Hh * VHD * Sl * 2);
    float* attn_f = (float*)alloc((size_t)ROWS * Hh * VHD * 4);
    unsigned short* attn_b = (unsigned short*)alloc((size_t)ROWS * Hh * VHD * 2);

    dim3 blk(256);
    dim3 tblk(32, 8);
    int gm = ROWS / GTM; // 32

    // operand prep: bf16 convert + weight transposes
    conv_bf16_kernel<<<2048, blk, 0, stream>>>(hidden, hid_b, ROWS, INs, INs);
    transpose_bf16_kernel<<<dim3(QR / 32, INs / 32), tblk, 0, stream>>>(Wq_a, WqaT, INs, QR, QR);
    transpose_bf16_kernel<<<dim3(Hh * QKD / 32, QR / 32), tblk, 0, stream>>>(Wq_b, WqbT, QR, Hh * QKD, Hh * QKD);
    transpose_bf16_kernel<<<dim3(KVCP / 32, INs / 32), tblk, 0, stream>>>(Wkv_a, WkvaT, INs, KVR + ROPE, KVCP);
    transpose_bf16_kernel<<<dim3(Hh * (NOPE + VHD) / 32, KVR / 32), tblk, 0, stream>>>(Wkv_b, WkvbT, KVR, Hh * (NOPE + VHD), Hh * (NOPE + VHD));
    transpose_bf16_kernel<<<dim3(HID / 32, (Hh * VHD) / 32), tblk, 0, stream>>>(Wo, WoT, Hh * VHD, HID, HID);

    // 1) q_a = hidden @ Wq_a
    gemm_bf16_kernel<<<dim3(QR / GTN, gm), blk, 0, stream>>>(hid_b, WqaT, qa_f, ROWS, QR, INs);
    // 2) rms(q_a) + convert
    rms_kernel<<<ROWS, blk, 0, stream>>>(qa_f, q_ln, QR, QR);
    conv_bf16_kernel<<<2048, blk, 0, stream>>>(qa_f, qa_b, ROWS, QR, QR);
    // 3) q = q_a_n @ Wq_b
    gemm_bf16_kernel<<<dim3(Hh * QKD / GTN, gm), blk, 0, stream>>>(qa_b, WqbT, q_f, ROWS, Hh * QKD, QR);
    // 4) kvc = hidden @ Wkv_a (padded to 640 cols; cols 576..639 are exact zeros)
    gemm_bf16_kernel<<<dim3(KVCP / GTN, gm), blk, 0, stream>>>(hid_b, WkvaT, kvc_f, ROWS, KVCP, INs);
    // 5) rms(kv_comp) in place + convert
    rms_kernel<<<ROWS, blk, 0, stream>>>(kvc_f, kv_ln, KVR, KVCP);
    conv_bf16_kernel<<<2048, blk, 0, stream>>>(kvc_f, kvn_b, ROWS, KVR, KVCP);
    // 6) kv = kv_comp_n @ Wkv_b
    gemm_bf16_kernel<<<dim3(Hh * (NOPE + VHD) / GTN, gm), blk, 0, stream>>>(kvn_b, WkvbT, kv_f, ROWS, Hh * (NOPE + VHD), KVR);
    // 7) RoPE + pack bf16 Q/K/V, then V transpose to [B,H,VHD,S]
    rope_pack_kernel<<<ROWS, blk, 0, stream>>>(q_f, kv_f, kvc_f, pos, Qf, Kf, Vb);
    transpose_v_kernel<<<dim3(VHD / 32, Sl / 32, Bsz * Hh), tblk, 0, stream>>>(Vb, Vtg);
    // 8) causal flash attention
    mla_attn_kernel<<<dim3(Sl / 64, Hh, Bsz), dim3(128), 0, stream>>>(Qf, Kf, Vtg, attn_f);
    // 9) out = attn @ Wo
    conv_bf16_kernel<<<2048, blk, 0, stream>>>(attn_f, attn_b, ROWS, Hh * VHD, Hh * VHD);
    gemm_bf16_kernel<<<dim3(HID / GTN, gm), blk, 0, stream>>>(attn_b, WoT, (float*)d_out, ROWS, HID, Hh * VHD);
}
